// Quantization_76776835383753
// MI455X (gfx1250) — matrix operand order repaired and checked by reading, not compiled, not run on hardware
//
#include <hip/hip_runtime.h>
#include <stdint.h>

#define N_TOK 32768
#define DIMS  512
#define KCB   8192
#define BM    128
#define BN    128
#define INV_TEMP 0.8f   // 1/1.25

typedef __attribute__((ext_vector_type(4)))  __bf16 v4bf;
typedef __attribute__((ext_vector_type(8)))  __bf16 v8bf;
typedef __attribute__((ext_vector_type(16))) __bf16 v16bf;
typedef __attribute__((ext_vector_type(8)))  float  v8f;

#define A_STRIDE 520   // 512 + 8 pad (16B aligned rows, conflict-free frag reads)
#define B_STRIDE 72    // 64 + 8 pad

// CDNA5 async global->LDS path (ASYNCcnt), guarded so absence falls back to
// plain LDS stores without breaking the build.
#if __has_builtin(__builtin_amdgcn_global_load_async_to_lds_b128) && \
    __has_builtin(__builtin_amdgcn_s_wait_asynccnt)
#define USE_ASYNC 1
#define WAIT_ASYNC(n) __builtin_amdgcn_s_wait_asynccnt(n)
#else
#define USE_ASYNC 0
#define WAIT_ASYNC(n) ((void)0)
#endif

// Builtin parameter types (from hipcc diagnostic): GCC-style int4 vectors,
// global (AS1) source pointer and LDS (AS3) destination pointer.
typedef int v4i_gcc __attribute__((vector_size(16)));
typedef __attribute__((address_space(1))) v4i_gcc* gptr4_t;
typedef __attribute__((address_space(3))) v4i_gcc* lptr4_t;

__device__ __forceinline__ uint32_t mix32(uint32_t a, uint32_t b) {
    uint32_t x = a * 0x85ebca6bu ^ (b + 0x9e3779b9u + (a << 6) + (a >> 2));
    x ^= x >> 16; x *= 0x7feb352du;
    x ^= x >> 15; x *= 0x846ca68bu;
    x ^= x >> 16;
    return x;
}

// Deterministic counter-based Gumbel noise for (row, code), seed 42.
__device__ __forceinline__ float gumbel_noise(uint32_t row, uint32_t k) {
    uint32_t h = mix32(row * 0x01000193u + 42u, k);
    float u = ((float)(h >> 8) + 0.5f) * (1.0f / 16777216.0f);   // (0,1)
    return -__logf(-__logf(u));
}

__device__ __forceinline__ float wave_sum(float v) {
    #pragma unroll
    for (int m = 16; m >= 1; m >>= 1) v += __shfl_xor(v, m, 32);
    return v;
}

// ---------------------------------------------------------------------------
// Kernel 1: codebook squared norms + f32->bf16 codebook conversion (done once
// so the GEMM streams bf16 from L2) + zero the loss accumulator.
// ---------------------------------------------------------------------------
__global__ __launch_bounds__(256)
void vq_prep_kernel(const float* __restrict__ cb,
                    float* __restrict__ cnorm,
                    __bf16* __restrict__ cbf16,
                    float* __restrict__ loss_out) {
    const int tid  = threadIdx.x;
    const int wave = tid >> 5;
    const int lane = tid & 31;
    const int code = blockIdx.x * 8 + wave;

    const float4* c4 = reinterpret_cast<const float4*>(cb);
    float s = 0.0f;
    #pragma unroll
    for (int j = 0; j < 4; ++j) {
        const int cidx = lane + 32 * j;
        float4 v = c4[(size_t)code * (DIMS / 4) + cidx];
        s += v.x * v.x + v.y * v.y + v.z * v.z + v.w * v.w;
        v4bf b;
        b[0] = (__bf16)v.x; b[1] = (__bf16)v.y;
        b[2] = (__bf16)v.z; b[3] = (__bf16)v.w;
        *reinterpret_cast<v4bf*>(&cbf16[(size_t)code * DIMS + 4 * cidx]) = b;
    }
    s = wave_sum(s);
    if (lane == 0) cnorm[code] = s;
    if (blockIdx.x == 0 && tid == 0) *loss_out = 0.0f;
}

// Stage one B tile: codes [kc,kc+128) x dims [d0,d0+64), bf16 -> LDS.
// 1024 x b128 transfers, 4 per thread. Async path goes straight to LDS.
__device__ __forceinline__ void stage_B(const __bf16* __restrict__ cbf16,
                                        __bf16* blds, int kc, int d0, int tid) {
    #pragma unroll
    for (int it = 0; it < 4; ++it) {
        const int i  = tid + it * 256;
        const int e  = i << 3;         // bf16 element index (8 per b128)
        const int cr = e >> 6;         // code within chunk
        const int c  = e & 63;         // dim within chunk
        const __bf16* src = &cbf16[(size_t)(kc + cr) * DIMS + d0 + c];
        __bf16* dst = &blds[cr * B_STRIDE + c];
#if USE_ASYNC
        __builtin_amdgcn_global_load_async_to_lds_b128(
            (gptr4_t)(uintptr_t)src,
            (lptr4_t)(uint32_t)(uintptr_t)dst, 0, 0);
#else
        *reinterpret_cast<v8bf*>(dst) = *reinterpret_cast<const v8bf*>(src);
#endif
    }
}

// Load one 16-bf16 B fragment (two 16B LDS reads at D offsets 0 and 16).
__device__ __forceinline__ v16bf load_bfrag(const __bf16* bb) {
    v8bf blo = *reinterpret_cast<const v8bf*>(bb);
    v8bf bhi = *reinterpret_cast<const v8bf*>(bb + 16);
    return __builtin_shufflevector(blo, bhi,
           0,1,2,3,4,5,6,7,8,9,10,11,12,13,14,15);
}

// ---------------------------------------------------------------------------
// Kernel 2: fused bf16-WMMA GEMM (x @ codebook^T) + Gumbel-max argmax per row.
//   score(n,k) = (2*x.c_k - ||c_k||^2)/TEMP + gumbel(n,k)
// Block = 256 thr (8 waves). x rows staged once in LDS (bf16, full D, 130 KB —
// needs CDNA5's 320KB WGP LDS); bf16 codebook streamed through a
// double-buffered LDS tile via async global->LDS copies overlapped with WMMA.
// B fragments are software-pipelined (depth 1) so a ds_load is always in
// flight while the previous WMMA executes.
// ---------------------------------------------------------------------------
__global__ __launch_bounds__(256)
void vq_score_kernel(const float* __restrict__ x,
                     const __bf16* __restrict__ cbf16,
                     const float* __restrict__ cnorm,
                     int* __restrict__ ids) {
    __shared__ __bf16 Alds[BM * A_STRIDE];        // 133,120 B
    __shared__ __bf16 Blds[2][BN * B_STRIDE];     //  36,864 B

    const int tid   = threadIdx.x;
    const int wave  = tid >> 5;
    const int lane  = tid & 31;
    const int nloc  = lane & 15;
    const bool hi   = lane >= 16;
    const int row_base = blockIdx.x * BM;

    const float4* x4 = reinterpret_cast<const float4*>(x);

    // ---- stage A: 128 rows x 512 dims, f32 -> bf16, once per block ----
    for (int i = tid; i < BM * DIMS / 4; i += 256) {
        const int e = i << 2;
        const int r = e >> 9;
        const int c = e & (DIMS - 1);
        float4 v = x4[(size_t)(row_base + r) * (DIMS / 4) + (c >> 2)];
        v4bf b;
        b[0] = (__bf16)v.x; b[1] = (__bf16)v.y;
        b[2] = (__bf16)v.z; b[3] = (__bf16)v.w;
        *reinterpret_cast<v4bf*>(&Alds[r * A_STRIDE + c]) = b;
    }

    float best[8];
    int   bidx[8];
    #pragma unroll
    for (int r = 0; r < 8; ++r) { best[r] = -3.4e38f; bidx[r] = 0; }

    // A-fragment addressing (16-bit A 16x32 layout):
    //  lane<16 : M=lane,   K in {0..7, 16..23}
    //  lane>=16: M=lane-16,K in {8..15,24..31}
    const int mrow = wave * 16 + nloc;
    const int ao   = hi ? 8 : 0;
    const int bo   = hi ? 8 : 0;

    const int NTILE = (KCB / BN) * (DIMS / 64);   // 512 tiles of 128 codes x 64 dims

    // prologue: kick off tile 0 into buffer 0
    stage_B(cbf16, &Blds[0][0], 0, 0, tid);

    v8f acc[8];
    const v8f vzero = {};
    #pragma unroll
    for (int t = 0; t < 8; ++t) acc[t] = vzero;

    int sel = 0;
    for (int tile = 0; tile < NTILE; ++tile) {
        const int kc = (tile >> 3) * BN;
        const int d0 = (tile & 7) * 64;

        // all waves done reading buf[sel^1] (last read during the previous tile)
        __syncthreads();
        if (tile + 1 < NTILE) {
            const int nt = tile + 1;
            stage_B(cbf16, &Blds[sel ^ 1][0], (nt >> 3) * BN, (nt & 7) * 64, tid);
            WAIT_ASYNC(4);     // in-order: oldest 4 (current tile) have landed
        } else {
            WAIT_ASYNC(0);
        }
        __syncthreads();        // current buffer visible to all waves

        const __bf16* bbase = &Blds[sel][0] + nloc * B_STRIDE + bo;
        #pragma unroll
        for (int dc = 0; dc < 64; dc += 32) {
            const __bf16* ab = &Alds[mrow * A_STRIDE + d0 + dc + ao];
            v8bf alo = *reinterpret_cast<const v8bf*>(ab);
            v8bf ahi = *reinterpret_cast<const v8bf*>(ab + 16);
            v16bf a = __builtin_shufflevector(alo, ahi,
                      0,1,2,3,4,5,6,7,8,9,10,11,12,13,14,15);
            // depth-1 software pipeline: B(t+1) load issued before WMMA(t),
            // so each WMMA waits only on a load that has had a WMMA's worth
            // of latency to complete (s_wait_dscnt <= 1 instead of 0).
            v16bf bcur = load_bfrag(bbase + dc);
            #pragma unroll
            for (int t = 0; t < 8; ++t) {
                v16bf bnxt = bcur;
                if (t < 7)
                    bnxt = load_bfrag(bbase + (t + 1) * (16 * B_STRIDE) + dc);
                acc[t] = __builtin_amdgcn_wmma_f32_16x16x32_bf16(
                             false, a, false, bcur, (short)0, acc[t],
                             false, false);
                bcur = bnxt;
            }
        }
        sel ^= 1;

        if ((tile & 7) == 7) {
            // ---- epilogue: full-D dot products done for codes [kc,kc+128) ----
            // C layout: lanes 0-15 VGPR r -> (M=r, N=lane); lanes 16-31 -> M=r+8.
            #pragma unroll
            for (int t = 0; t < 8; ++t) {
                const int k  = kc + t * 16 + nloc;
                const float cn = cnorm[k];
                #pragma unroll
                for (int r = 0; r < 8; ++r) {
                    const int row = row_base + wave * 16 + (hi ? 8 : 0) + r;
                    float s = fmaf(2.0f, acc[t][r], -cn) * INV_TEMP
                            + gumbel_noise((uint32_t)row, (uint32_t)k);
                    if (s > best[r]) { best[r] = s; bidx[r] = k; }
                }
                acc[t] = vzero;
            }
        }
    }

    // ---- cross-lane argmax within each 16-lane half ----
    #pragma unroll
    for (int r = 0; r < 8; ++r) {
        #pragma unroll
        for (int m = 8; m >= 1; m >>= 1) {
            float os = __shfl_xor(best[r], m, 32);
            int   oi = __shfl_xor(bidx[r], m, 32);
            if (os > best[r]) { best[r] = os; bidx[r] = oi; }
        }
        if (lane == 0)  ids[row_base + wave * 16 + r]     = bidx[r];
        if (lane == 16) ids[row_base + wave * 16 + 8 + r] = bidx[r];
    }
}

// ---------------------------------------------------------------------------
// Kernel 3: gather + rotation-trick transform + loss. One row per wave.
// ---------------------------------------------------------------------------
__global__ __launch_bounds__(256)
void vq_rotate_kernel(const float* __restrict__ x,
                      const float* __restrict__ cb,
                      const int* __restrict__ ids,
                      float* __restrict__ emb_out,
                      float* __restrict__ ids_out,
                      float* __restrict__ loss_out) {
    __shared__ float blk_loss;
    const int tid  = threadIdx.x;
    const int wave = tid >> 5;
    const int lane = tid & 31;
    const int row  = blockIdx.x * 8 + wave;
    if (tid == 0) blk_loss = 0.0f;
    __syncthreads();

    const int id = ids[row];
    const float4* x4 = reinterpret_cast<const float4*>(x);
    const float4* c4 = reinterpret_cast<const float4*>(cb);

    float4 xv[4], ev[4];
    float sx2 = 0.0f, se2 = 0.0f, sxe = 0.0f;
    #pragma unroll
    for (int j = 0; j < 4; ++j) {
        const int cidx = lane + 32 * j;
        xv[j] = x4[(size_t)row * (DIMS / 4) + cidx];
        ev[j] = c4[(size_t)id  * (DIMS / 4) + cidx];
        sx2 += xv[j].x*xv[j].x + xv[j].y*xv[j].y + xv[j].z*xv[j].z + xv[j].w*xv[j].w;
        se2 += ev[j].x*ev[j].x + ev[j].y*ev[j].y + ev[j].z*ev[j].z + ev[j].w*ev[j].w;
        sxe += xv[j].x*ev[j].x + xv[j].y*ev[j].y + xv[j].z*ev[j].z + xv[j].w*ev[j].w;
    }
    sx2 = wave_sum(sx2);
    se2 = wave_sum(se2);
    sxe = wave_sum(sxe);

    const float nx = sqrtf(sx2), ne = sqrtf(se2);
    const float iu = 1.0f / (nx + 1e-8f);      // u = x * iu
    const float iq = 1.0f / (ne + 1e-8f);      // q = e * iq
    const float xu = sx2 * iu;                 // x.u
    const float xq = sxe * iq;                 // x.q
    const float nw2 = sx2 * iu * iu + se2 * iq * iq + 2.0f * sxe * iu * iq;
    const float inw = 1.0f / fmaxf(sqrtf(nw2), 1e-6f);
    const float xw  = (xu + xq) * inw;         // x.w
    const float cw  = 2.0f * xw * inw;         // folds w = (u+q)*inw
    const float cq  = 2.0f * xu * iq;

    float4* o4 = reinterpret_cast<float4*>(emb_out);
    #pragma unroll
    for (int j = 0; j < 4; ++j) {
        float4 o;
        o.x = xv[j].x - cw * (xv[j].x * iu + ev[j].x * iq) + cq * ev[j].x;
        o.y = xv[j].y - cw * (xv[j].y * iu + ev[j].y * iq) + cq * ev[j].y;
        o.z = xv[j].z - cw * (xv[j].z * iu + ev[j].z * iq) + cq * ev[j].z;
        o.w = xv[j].w - cw * (xv[j].w * iu + ev[j].w * iq) + cq * ev[j].w;
        o4[(size_t)row * (DIMS / 4) + lane + 32 * j] = o;
    }

    if (lane == 0) {
        ids_out[row] = (float)id;
        // loss = sum((x-e)^2) + 0.25*sum((x-e)^2) = 1.25*(sx2 - 2 sxe + se2)
        atomicAdd(&blk_loss, 1.25f * (sx2 - 2.0f * sxe + se2));
    }
    __syncthreads();
    if (tid == 0) atomicAdd(loss_out, blk_loss);
}

// ---------------------------------------------------------------------------
extern "C" void kernel_launch(void* const* d_in, const int* in_sizes, int n_in,
                              void* d_out, int out_size, void* d_ws, size_t ws_size,
                              hipStream_t stream) {
    const float* x  = (const float*)d_in[0];   // [32768, 512]
    const float* cb = (const float*)d_in[1];   // [8192, 512]

    float* out      = (float*)d_out;
    float* emb_out  = out;                               // [N, D]
    float* ids_out  = out + (size_t)N_TOK * DIMS;        // [N]
    float* loss_out = ids_out + N_TOK;                   // scalar

    // workspace: cnorm (32KB) | ids (128KB) | bf16 codebook (8MB)
    float*  cnorm = (float*)d_ws;
    int*    ids   = (int*)((char*)d_ws + KCB * sizeof(float));
    __bf16* cbf16 = (__bf16*)((char*)d_ws + KCB * sizeof(float)
                                          + N_TOK * sizeof(int));

    vq_prep_kernel  <<<KCB / 8,    256, 0, stream>>>(cb, cnorm, cbf16, loss_out);
    vq_score_kernel <<<N_TOK / BM, 256, 0, stream>>>(x, cbf16, cnorm, ids);
    vq_rotate_kernel<<<N_TOK / 8,  256, 0, stream>>>(x, cb, ids, emb_out, ids_out, loss_out);
}
